// TransformerEncoderLayer_71691594105101
// MI455X (gfx1250) — matrix operand order repaired and checked by reading, not compiled, not run on hardware
//
#include <hip/hip_runtime.h>
#include <cstdint>
#include <cstddef>

// ---------------------------------------------------------------------------
// Types / helpers
// ---------------------------------------------------------------------------
typedef __attribute__((ext_vector_type(16))) __bf16 v16bf;
typedef __attribute__((ext_vector_type(8)))  float  v8f;

union FragU { uint4 q[2]; v16bf v; };

#if defined(__AMDGCN__) && __has_builtin(__builtin_amdgcn_global_load_async_to_lds_b128)
#define HAVE_ASYNC_LDS 1
#endif

typedef int v4i __attribute__((vector_size(16)));
typedef __attribute__((address_space(1))) v4i gv4i_t;
typedef __attribute__((address_space(3))) v4i lv4i_t;

__device__ __forceinline__ void async_copy16(const unsigned short* g, unsigned short* l) {
#ifdef HAVE_ASYNC_LDS
  __builtin_amdgcn_global_load_async_to_lds_b128((gv4i_t*)g, (lv4i_t*)l, 0, 0);
#else
  (void)g; (void)l;
#endif
}
__device__ __forceinline__ void wait_async0() {
#if defined(__AMDGCN__) && __has_builtin(__builtin_amdgcn_s_wait_asynccnt)
  __builtin_amdgcn_s_wait_asynccnt(0);
#elif defined(__AMDGCN__)
  asm volatile("s_wait_asynccnt 0" ::: "memory");
#endif
}

__device__ __forceinline__ float bf2f(unsigned short h) {
  union { unsigned int u; float f; } c; c.u = ((unsigned int)h) << 16; return c.f;
}
__device__ __forceinline__ unsigned short f2bf(float f) {
  union { float f; unsigned int u; } c; c.f = f;
  unsigned int u = c.u;
  unsigned int r = (u + 0x7FFFu + ((u >> 16) & 1u)) >> 16;
  return (unsigned short)r;
}
__device__ __forceinline__ float gelu_exact(float x) {
  return 0.5f * x * (1.0f + erff(x * 0.70710678118654752f));
}

// Problem constants
#define DMODEL 1536
#define FFN_D  6144
#define ADH    384
#define NTOK   257
#define BT_    64
#define TOKENS 16448     // 257 * 64
#define NHEAD  24
#define HD_    64
#define FFN_CHUNK 2056   // 16448 / 8

// ---------------------------------------------------------------------------
// bf16 GEMM with WMMA:  out = act((A @ B + bias) * scale) [+ old out]
// A : [M][K]  bf16 row-major (activations)
// BT: [N][K]  bf16 row-major (weights, PRE-TRANSPOSED once per launch)
// Block tile 128(M) x 128(N), K-step 32, 8 waves (wave32):
//   wave grid 4(M) x 2(N), wave tile 32 x 64 = 2x4 WMMA accumulators.
// Double-buffered LDS (2 x (8KB A + 8KB B) = 32 KB), one barrier per K-step.
// Staging uses GLOBAL_LOAD_ASYNC_TO_LDS_B128 when the toolchain exposes it,
// else coalesced b128 loads through VGPRs.
// flags: bit0 = exact GELU, bit1 = accumulate into outF.
// ---------------------------------------------------------------------------
struct GArgs {
  const unsigned short* A;
  const unsigned short* BT;
  const float* bias;
  float* outF;
  unsigned short* outH;
  int M, N, K;
  float scale;
  int flags;
};

__global__ __launch_bounds__(256) void gemm_bf16_wmma(GArgs g) {
  __shared__ __align__(16) unsigned short As[2][128 * 32];
  __shared__ __align__(16) unsigned short Bs[2][128 * 32];   // [n][k]

  const int tid  = threadIdx.x;
  const int lane = tid & 31;
  const int wave = tid >> 5;
  const int wm   = wave & 3;   // 4 waves along M (x32)
  const int wn   = wave >> 2;  // 2 waves along N (x64)
  const int bm   = blockIdx.x * 128;
  const int bn   = blockIdx.y * 128;

  v8f acc[2][4];
  const v8f vzero = {0.f, 0.f, 0.f, 0.f, 0.f, 0.f, 0.f, 0.f};
#pragma unroll
  for (int i = 0; i < 2; ++i)
#pragma unroll
    for (int j = 0; j < 4; ++j) acc[i][j] = vzero;

  // staging assignment: each thread covers 16 halves (32 B) of A and of B^T
  const int srow = tid >> 1;          // 0..127
  const int scol = (tid & 1) * 16;    // 0 or 16 (halves)
  const int lhalf = (lane < 16) ? 1 : 0;
  const int aguard = (bm + srow) < g.M;

  const unsigned short* Aptr = g.A  + (size_t)(bm + srow) * g.K + scol;
  const unsigned short* Bptr = g.BT + (size_t)(bn + srow) * g.K + scol;

  const int S = g.K >> 5;   // K / 32, always >= 2 here

#ifdef HAVE_ASYNC_LDS
  // Pre-zero M-tail rows in both buffers (async loads are lane-predicated).
  if (!aguard) {
    const uint4 z = make_uint4(0u, 0u, 0u, 0u);
    *((uint4*)&As[0][srow * 32 + scol])     = z;
    *((uint4*)&As[0][srow * 32 + scol + 8]) = z;
    *((uint4*)&As[1][srow * 32 + scol])     = z;
    *((uint4*)&As[1][srow * 32 + scol + 8]) = z;
  }
  // ---- stage tile 0 asynchronously ----
  if (aguard) {
    async_copy16(Aptr,     &As[0][srow * 32 + scol]);
    async_copy16(Aptr + 8, &As[0][srow * 32 + scol + 8]);
  }
  async_copy16(Bptr,     &Bs[0][srow * 32 + scol]);
  async_copy16(Bptr + 8, &Bs[0][srow * 32 + scol + 8]);
  wait_async0();
  __syncthreads();
#else
  uint4 a0, a1, b0, b1;
  // ---- stage tile 0 ----
  {
    a0 = make_uint4(0u, 0u, 0u, 0u); a1 = a0;
    if (aguard) {
      const uint4* p = (const uint4*)Aptr;
      a0 = p[0]; a1 = p[1];
    }
    const uint4* q = (const uint4*)Bptr;
    b0 = q[0]; b1 = q[1];
    *((uint4*)&As[0][srow * 32 + scol])     = a0;
    *((uint4*)&As[0][srow * 32 + scol + 8]) = a1;
    *((uint4*)&Bs[0][srow * 32 + scol])     = b0;
    *((uint4*)&Bs[0][srow * 32 + scol + 8]) = b1;
  }
  __syncthreads();
#endif

  for (int s = 0; s < S; ++s) {
    const int cur = s & 1;
    const int nxt = cur ^ 1;
    const bool more = (s + 1) < S;

    if (more) {
      const size_t koff = (size_t)(s + 1) * 32;
#ifdef HAVE_ASYNC_LDS
      // buffer `nxt` was last read in iteration s-1 (barrier since) -> safe
      if (aguard) {
        async_copy16(Aptr + koff,     &As[nxt][srow * 32 + scol]);
        async_copy16(Aptr + koff + 8, &As[nxt][srow * 32 + scol + 8]);
      }
      async_copy16(Bptr + koff,     &Bs[nxt][srow * 32 + scol]);
      async_copy16(Bptr + koff + 8, &Bs[nxt][srow * 32 + scol + 8]);
#else
      a0 = make_uint4(0u, 0u, 0u, 0u); a1 = a0;
      if (aguard) {
        const uint4* p = (const uint4*)(Aptr + koff);
        a0 = p[0]; a1 = p[1];
      }
      const uint4* q = (const uint4*)(Bptr + koff);
      b0 = q[0]; b1 = q[1];
#endif
      if (s + 2 < S) {
        __builtin_prefetch(Aptr + koff + 32, 0, 1);
        __builtin_prefetch(Bptr + koff + 32, 0, 1);
      }
    }

    // ---- compute on buffer `cur` ----
    const unsigned short* Ab = &As[cur][0];
    const unsigned short* Bb = &Bs[cur][0];
#pragma unroll
    for (int mi = 0; mi < 2; ++mi) {
      const int rl    = wm * 32 + mi * 16 + (lane & 15);
      const int abase = lhalf ? 0 : 8;
      FragU af;
      af.q[0] = *((const uint4*)&Ab[rl * 32 + abase]);
      af.q[1] = *((const uint4*)&Ab[rl * 32 + abase + 16]);
#pragma unroll
      for (int ni = 0; ni < 4; ++ni) {
        const int cl = wn * 64 + ni * 16 + (lane & 15);
        const int kb = lhalf ? 0 : 8;
        FragU bf;
        bf.q[0] = *((const uint4*)&Bb[cl * 32 + kb]);
        bf.q[1] = *((const uint4*)&Bb[cl * 32 + kb + 16]);
        acc[mi][ni] = __builtin_amdgcn_wmma_f32_16x16x32_bf16(
            false, af.v, false, bf.v, (short)0, acc[mi][ni], false, false);
      }
    }

    if (more) {
#ifdef HAVE_ASYNC_LDS
      wait_async0();
      __syncthreads();
#else
      *((uint4*)&As[nxt][srow * 32 + scol])     = a0;
      *((uint4*)&As[nxt][srow * 32 + scol + 8]) = a1;
      *((uint4*)&Bs[nxt][srow * 32 + scol])     = b0;
      *((uint4*)&Bs[nxt][srow * 32 + scol + 8]) = b1;
      __syncthreads();
#endif
    }
  }

  // ---- epilogue ----
#pragma unroll
  for (int mi = 0; mi < 2; ++mi) {
#pragma unroll
    for (int ni = 0; ni < 4; ++ni) {
      const int gn = bn + wn * 64 + ni * 16 + (lane & 15);
      const float bv = g.bias ? g.bias[gn] : 0.0f;
#pragma unroll
      for (int r = 0; r < 8; ++r) {
        const int gm = bm + wm * 32 + mi * 16 + r + (lhalf ? 0 : 8);
        if (gm >= g.M) continue;
        float v = (acc[mi][ni][r] + bv) * g.scale;
        if (g.flags & 1) v = gelu_exact(v);
        const size_t idx = (size_t)gm * g.N + gn;
        if (g.flags & 2) v += g.outF[idx];
        if (g.outF) g.outF[idx] = v;
        if (g.outH) g.outH[idx] = f2bf(v);
      }
    }
  }
}

// ---------------------------------------------------------------------------
// Weight convert + transpose:  W[K][N] f32  ->  WT[N][K] bf16
// ---------------------------------------------------------------------------
__global__ __launch_bounds__(256) void f2bf_transpose_kernel(const float* __restrict__ W,
                                                             unsigned short* __restrict__ WT,
                                                             int K, int N) {
  __shared__ unsigned short t[32][33];
  const int n0 = blockIdx.x * 32;
  const int k0 = blockIdx.y * 32;
  const int tx = threadIdx.x;   // 0..31
  const int ty = threadIdx.y;   // 0..7
#pragma unroll
  for (int j = 0; j < 4; ++j)
    t[ty + 8 * j][tx] = f2bf(W[(size_t)(k0 + ty + 8 * j) * N + (n0 + tx)]);
  __syncthreads();
#pragma unroll
  for (int j = 0; j < 4; ++j)
    WT[(size_t)(n0 + ty + 8 * j) * K + (k0 + tx)] = t[tx][ty + 8 * j];
}

// ---------------------------------------------------------------------------
// LayerNorm over last dim D (per-row block). src f32; optional f32/bf16 outs.
// ---------------------------------------------------------------------------
__global__ __launch_bounds__(256) void ln_kernel(const float* __restrict__ src,
                                                 const float* __restrict__ w,
                                                 const float* __restrict__ b,
                                                 float* outF, unsigned short* outH,
                                                 int D) {
  const int row = blockIdx.x;
  const float* x = src + (size_t)row * D;
  __shared__ float red[256];
  const int tid = threadIdx.x;

  float s = 0.f, s2 = 0.f;
  for (int i = tid; i < D; i += 256) { const float v = x[i]; s += v; s2 += v * v; }

  red[tid] = s; __syncthreads();
  for (int off = 128; off > 0; off >>= 1) {
    if (tid < off) red[tid] += red[tid + off];
    __syncthreads();
  }
  const float mean = red[0] / (float)D;
  __syncthreads();

  red[tid] = s2; __syncthreads();
  for (int off = 128; off > 0; off >>= 1) {
    if (tid < off) red[tid] += red[tid + off];
    __syncthreads();
  }
  const float var  = red[0] / (float)D - mean * mean;
  const float rinv = rsqrtf(var + 1e-5f);

  for (int i = tid; i < D; i += 256) {
    const float v = (x[i] - mean) * rinv * w[i] + b[i];
    const size_t idx = (size_t)row * D + i;
    if (outF) outF[idx] = v;
    if (outH) outH[idx] = f2bf(v);
  }
}

// ---------------------------------------------------------------------------
// Temporal attention: groups (b, n, h), L = T = 8.  tok(t) = n*64 + b*8 + t.
// ---------------------------------------------------------------------------
__global__ __launch_bounds__(64) void attn_temporal(const unsigned short* __restrict__ Q,
                                                    const unsigned short* __restrict__ K,
                                                    const unsigned short* __restrict__ V,
                                                    float* __restrict__ O) {
  const int gid = blockIdx.x;          // 2056 * 24 groups
  const int h  = gid % NHEAD;
  const int bn = gid / NHEAD;          // 0..2055
  const int b  = bn & 7;
  const int n  = bn >> 3;

  __shared__ float qs[8][64], ks[8][64], vs[8][64];
  __shared__ float sc[8][9];
  const int tid = threadIdx.x;

  for (int t = 0; t < 8; ++t) {
    const size_t base = ((size_t)(n * 64 + b * 8 + t)) * DMODEL + h * HD_ + tid;
    qs[t][tid] = bf2f(Q[base]);
    ks[t][tid] = bf2f(K[base]);
    vs[t][tid] = bf2f(V[base]);
  }
  __syncthreads();

  const int tq = tid >> 3, tk = tid & 7;
  float s = 0.f;
  for (int d = 0; d < 64; ++d) s += qs[tq][d] * ks[tk][d];
  sc[tq][tk] = s;
  __syncthreads();

  if (tid < 8) {
    float m = sc[tid][0];
    for (int j = 1; j < 8; ++j) m = fmaxf(m, sc[tid][j]);
    float sum = 0.f;
    for (int j = 0; j < 8; ++j) { const float e = __expf(sc[tid][j] - m); sc[tid][j] = e; sum += e; }
    const float inv = 1.f / sum;
    for (int j = 0; j < 8; ++j) sc[tid][j] *= inv;
  }
  __syncthreads();

  const int db = (tid & 7) * 8;
  for (int dd = 0; dd < 8; ++dd) {
    const int d = db + dd;
    float o = 0.f;
    for (int j = 0; j < 8; ++j) o += sc[tq][j] * vs[j][d];
    O[((size_t)(n * 64 + b * 8 + tq)) * DMODEL + h * HD_ + d] = o;
  }
}

// ---------------------------------------------------------------------------
// Spatial attention with relative-position bias: groups (bt, h), L = 257.
// tok(j) = j*64 + bt.  One block per (query qi, head h, batch bt).
// ---------------------------------------------------------------------------
__global__ __launch_bounds__(256) void attn_spatial(const unsigned short* __restrict__ Q,
                                                    const unsigned short* __restrict__ K,
                                                    const unsigned short* __restrict__ V,
                                                    const float* __restrict__ bias,
                                                    float* __restrict__ O) {
  const int qi = blockIdx.x;   // 0..256
  const int h  = blockIdx.y;   // 0..23
  const int bt = blockIdx.z;   // 0..63

  __shared__ float qv[64];
  __shared__ float s[NTOK];
  __shared__ float red[256];
  __shared__ float part[4][64];
  const int tid = threadIdx.x;

  if (tid < 64) qv[tid] = bf2f(Q[((size_t)(qi * 64 + bt)) * DMODEL + h * HD_ + tid]);
  __syncthreads();

  for (int j = tid; j < NTOK; j += 256) {
    const unsigned short* kp = K + ((size_t)(j * 64 + bt)) * DMODEL + h * HD_;
    float dot = 0.f;
#pragma unroll
    for (int c = 0; c < 8; ++c) {                 // 8 x uint4 = 64 bf16
      const uint4 d4 = *((const uint4*)(kp + c * 8));
      const unsigned short* us = (const unsigned short*)&d4;
#pragma unroll
      for (int e = 0; e < 8; ++e) dot += qv[c * 8 + e] * bf2f(us[e]);
    }
    s[j] = dot + bias[((size_t)h * NTOK + qi) * NTOK + j];
  }
  __syncthreads();

  float m = -1e30f;
  for (int j = tid; j < NTOK; j += 256) m = fmaxf(m, s[j]);
  red[tid] = m; __syncthreads();
  for (int off = 128; off > 0; off >>= 1) {
    if (tid < off) red[tid] = fmaxf(red[tid], red[tid + off]);
    __syncthreads();
  }
  m = red[0];
  __syncthreads();

  float ps = 0.f;
  for (int j = tid; j < NTOK; j += 256) { const float e = __expf(s[j] - m); s[j] = e; ps += e; }
  red[tid] = ps; __syncthreads();
  for (int off = 128; off > 0; off >>= 1) {
    if (tid < off) red[tid] += red[tid + off];
    __syncthreads();
  }
  const float inv = 1.f / red[0];
  __syncthreads();

  // AV: all 256 threads, 4-way split over keys, LDS partial reduction.
  {
    const int d   = tid & 63;
    const int grp = tid >> 6;   // 0..3
    float o = 0.f;
    for (int j = grp; j < NTOK; j += 4)
      o += s[j] * bf2f(V[((size_t)(j * 64 + bt)) * DMODEL + h * HD_ + d]);
    part[grp][d] = o;
    __syncthreads();
    if (tid < 64) {
      const float r = (part[0][tid] + part[1][tid]) + (part[2][tid] + part[3][tid]);
      O[((size_t)(qi * 64 + bt)) * DMODEL + h * HD_ + tid] = r * inv;
    }
  }
}

// ---------------------------------------------------------------------------
// Small elementwise / setup kernels (float4-vectorized where hot)
// ---------------------------------------------------------------------------
__global__ void bias_build_kernel(const float* __restrict__ relw, const int* __restrict__ bucket,
                                  float* __restrict__ out) {
  const int i = blockIdx.x * 256 + threadIdx.x;
  const int total = NHEAD * NTOK * NTOK;
  if (i >= total) return;
  const int h = i / (NTOK * NTOK);
  const int rest = i % (NTOK * NTOK);
  out[i] = relw[bucket[rest] * NHEAD + h];
}

__global__ void add4_kernel(const float4* __restrict__ a, const float4* __restrict__ b,
                            float4* __restrict__ o, int n4) {
  const int i = blockIdx.x * 256 + threadIdx.x;
  if (i < n4) {
    const float4 av = a[i], bv = b[i];
    o[i] = make_float4(av.x + bv.x, av.y + bv.y, av.z + bv.z, av.w + bv.w);
  }
}

// o = x + g[(4i..4i+3) % D] * y   (D % 4 == 0)
__global__ void gamma_add4_kernel(const float4* __restrict__ x, const float* __restrict__ g,
                                  const float4* __restrict__ y, float4* __restrict__ o,
                                  int n4, int D4) {
  const int i = blockIdx.x * 256 + threadIdx.x;
  if (i < n4) {
    const float4 gv = *((const float4*)(g + ((size_t)(i % D4) << 2)));
    const float4 xv = x[i], yv = y[i];
    o[i] = make_float4(xv.x + gv.x * yv.x, xv.y + gv.y * yv.y,
                       xv.z + gv.z * yv.z, xv.w + gv.w * yv.w);
  }
}

// a = gelu(a) * b
__global__ void gelu_mul4_kernel(float4* __restrict__ a, const float4* __restrict__ b, int n4) {
  const int i = blockIdx.x * 256 + threadIdx.x;
  if (i < n4) {
    const float4 av = a[i], bv = b[i];
    a[i] = make_float4(gelu_exact(av.x) * bv.x, gelu_exact(av.y) * bv.y,
                       gelu_exact(av.z) * bv.z, gelu_exact(av.w) * bv.w);
  }
}

// o = x2 + g2[..] * ffn + mbr
__global__ void final4_kernel(const float4* __restrict__ x2, const float* __restrict__ g2,
                              const float4* __restrict__ ffn, const float4* __restrict__ mbr,
                              float4* __restrict__ o, int n4, int D4) {
  const int i = blockIdx.x * 256 + threadIdx.x;
  if (i < n4) {
    const float4 gv = *((const float4*)(g2 + ((size_t)(i % D4) << 2)));
    const float4 xv = x2[i], fv = ffn[i], mv = mbr[i];
    o[i] = make_float4(xv.x + gv.x * fv.x + mv.x, xv.y + gv.y * fv.y + mv.y,
                       xv.z + gv.z * fv.z + mv.z, xv.w + gv.w * fv.w + mv.w);
  }
}

// ---------------------------------------------------------------------------
// Host orchestration
// ---------------------------------------------------------------------------
extern "C" void kernel_launch(void* const* d_in, const int* in_sizes, int n_in,
                              void* d_out, int out_size, void* d_ws, size_t ws_size,
                              hipStream_t stream) {
  (void)in_sizes; (void)n_in; (void)out_size; (void)ws_size;

  const float* x         = (const float*)d_in[0];
  const float* q_w       = (const float*)d_in[1];
  const float* k_w       = (const float*)d_in[2];
  const float* v_w       = (const float*)d_in[3];
  const float* out_w     = (const float*)d_in[4];
  const float* q_b       = (const float*)d_in[5];
  const float* v_b       = (const float*)d_in[6];
  const float* out_b     = (const float*)d_in[7];
  const float* attn_ln_w = (const float*)d_in[8];
  const float* attn_ln_b = (const float*)d_in[9];
  const float* sa_ln_w   = (const float*)d_in[10];
  const float* sa_ln_b   = (const float*)d_in[11];
  const float* fin_ln_w  = (const float*)d_in[12];
  const float* fin_ln_b  = (const float*)d_in[13];
  const float* wi0       = (const float*)d_in[14];
  const float* wi1       = (const float*)d_in[15];
  const float* ffn_ln_w  = (const float*)d_in[16];
  const float* ffn_ln_b  = (const float*)d_in[17];
  const float* ffn_out_w = (const float*)d_in[18];
  const float* ffn_out_b = (const float*)d_in[19];
  const float* gamma1    = (const float*)d_in[20];
  const float* gamma2    = (const float*)d_in[21];
  const float* t_fc1_w   = (const float*)d_in[22];
  const float* t_fc1_b   = (const float*)d_in[23];
  const float* t_fc2_w   = (const float*)d_in[24];
  const float* t_fc2_b   = (const float*)d_in[25];
  const float* s_fc1_w   = (const float*)d_in[26];
  const float* s_fc1_b   = (const float*)d_in[27];
  const float* s_fc2_w   = (const float*)d_in[28];
  const float* s_fc2_b   = (const float*)d_in[29];
  const float* m_fc1_w   = (const float*)d_in[30];
  const float* m_fc1_b   = (const float*)d_in[31];
  const float* m_fc2_w   = (const float*)d_in[32];
  const float* m_fc2_b   = (const float*)d_in[33];
  const float* rel_pos_w = (const float*)d_in[34];
  const int*   rp_bucket = (const int*)d_in[35];

  // ---- workspace layout ----
  size_t off = 0;
  auto wsa = [&](size_t bytes) -> char* {
    char* p = (char*)d_ws + off;
    off += (bytes + 255) & ~(size_t)255;
    return p;
  };
  const size_t WB_DD  = (size_t)DMODEL * DMODEL * 2;
  const size_t WB_DF  = (size_t)DMODEL * FFN_D * 2;
  const size_t WB_AD  = (size_t)DMODEL * ADH * 2;
  const size_t HBUF   = (size_t)TOKENS * DMODEL * 2;   // == FFN_CHUNK*FFN_D*4
  const size_t FBUF   = (size_t)TOKENS * DMODEL * 4;

  // all weights stored TRANSPOSED: WT[N][K] bf16
  unsigned short* WQ  = (unsigned short*)wsa(WB_DD);
  unsigned short* WK  = (unsigned short*)wsa(WB_DD);
  unsigned short* WV  = (unsigned short*)wsa(WB_DD);
  unsigned short* WO  = (unsigned short*)wsa(WB_DD);
  unsigned short* WI0 = (unsigned short*)wsa(WB_DF);
  unsigned short* WI1 = (unsigned short*)wsa(WB_DF);
  unsigned short* WFO = (unsigned short*)wsa(WB_DF);
  unsigned short* WT1 = (unsigned short*)wsa(WB_AD);
  unsigned short* WT2 = (unsigned short*)wsa(WB_AD);
  unsigned short* WS1 = (unsigned short*)wsa(WB_AD);
  unsigned short* WS2 = (unsigned short*)wsa(WB_AD);
  unsigned short* WM1 = (unsigned short*)wsa(WB_AD);
  unsigned short* WM2 = (unsigned short*)wsa(WB_AD);
  float*          BIAS= (float*)wsa((size_t)NHEAD * NTOK * NTOK * 4);
  unsigned short* H0  = (unsigned short*)wsa(HBUF);
  unsigned short* H1  = (unsigned short*)wsa(HBUF);
  unsigned short* H2  = (unsigned short*)wsa(HBUF);
  unsigned short* H3  = (unsigned short*)wsa(HBUF);
  float*          F0  = (float*)wsa(FBUF);
  float*          F1  = (float*)wsa(FBUF);
  float*          F2  = (float*)wsa(FBUF);

  auto convT = [&](const float* w, unsigned short* wt, int K, int N) {
    f2bf_transpose_kernel<<<dim3((unsigned)(N / 32), (unsigned)(K / 32)), dim3(32, 8), 0, stream>>>(
        w, wt, K, N);
  };
  auto gemm = [&](const unsigned short* A, const unsigned short* BT, const float* bias,
                  float* oF, unsigned short* oH, int M, int N, int K, float scale, int flags) {
    GArgs a; a.A = A; a.BT = BT; a.bias = bias; a.outF = oF; a.outH = oH;
    a.M = M; a.N = N; a.K = K; a.scale = scale; a.flags = flags;
    dim3 grid((unsigned)((M + 127) / 128), (unsigned)(N / 128));
    gemm_bf16_wmma<<<grid, dim3(256), 0, stream>>>(a);
  };
  auto ln = [&](const float* src, const float* w, const float* b, float* oF,
                unsigned short* oH, int rows, int D) {
    ln_kernel<<<dim3((unsigned)rows), dim3(256), 0, stream>>>(src, w, b, oF, oH, D);
  };

  const int NE  = TOKENS * DMODEL;                 // 25,264,128 elements
  const int NE4 = NE / 4;
  const unsigned NE4B = (unsigned)((NE4 + 255) / 256);
  const int D4 = DMODEL / 4;

  // 0) weights -> bf16 transposed ; relative-position bias table
  convT(q_w,       WQ,  DMODEL, DMODEL);
  convT(k_w,       WK,  DMODEL, DMODEL);
  convT(v_w,       WV,  DMODEL, DMODEL);
  convT(out_w,     WO,  DMODEL, DMODEL);
  convT(wi0,       WI0, DMODEL, FFN_D);
  convT(wi1,       WI1, DMODEL, FFN_D);
  convT(ffn_out_w, WFO, FFN_D,  DMODEL);
  convT(t_fc1_w,   WT1, DMODEL, ADH);
  convT(t_fc2_w,   WT2, ADH,    DMODEL);
  convT(s_fc1_w,   WS1, DMODEL, ADH);
  convT(s_fc2_w,   WS2, ADH,    DMODEL);
  convT(m_fc1_w,   WM1, DMODEL, ADH);
  convT(m_fc2_w,   WM2, ADH,    DMODEL);
  {
    const int total = NHEAD * NTOK * NTOK;
    bias_build_kernel<<<dim3((unsigned)((total + 255) / 256)), dim3(256), 0, stream>>>(
        rel_pos_w, rp_bucket, BIAS);
  }

  const float qscale = 0.125f;  // 64^-0.5

  // ---- temporal branch ----
  ln(x, sa_ln_w, sa_ln_b, nullptr, H0, TOKENS, DMODEL);                     // ln0 -> H0
  gemm(H0, WQ, q_b, nullptr, H1, TOKENS, DMODEL, DMODEL, qscale, 0);        // Q
  gemm(H0, WK, nullptr, nullptr, H2, TOKENS, DMODEL, DMODEL, 1.f, 0);       // K
  gemm(H0, WV, v_b, nullptr, H3, TOKENS, DMODEL, DMODEL, 1.f, 0);           // V
  attn_temporal<<<dim3(2056u * NHEAD), dim3(64), 0, stream>>>(H1, H2, H3, F0);
  ln(F0, attn_ln_w, attn_ln_b, nullptr, H1, TOKENS, DMODEL);                // attn_ln -> H1
  gemm(H1, WO, out_b, nullptr, H2, TOKENS, DMODEL, DMODEL, 1.f, 0);         // out proj -> H2 (bf16)
  gemm(H2, WT1, t_fc1_b, nullptr, H3, TOKENS, ADH, DMODEL, 1.f, 1);         // gelu fc1 -> H3
  gemm(H3, WT2, t_fc2_b, F0, nullptr, TOKENS, DMODEL, ADH, 1.f, 0);         // fc2 -> F0 (= xt)
  add4_kernel<<<dim3(NE4B), dim3(256), 0, stream>>>(
      (const float4*)x, (const float4*)F0, (float4*)F2, NE4);               // x1 = x + xt -> F2

  // ---- spatial branch ----
  ln(F2, sa_ln_w, sa_ln_b, nullptr, H0, TOKENS, DMODEL);                    // ln1 -> H0
  gemm(H0, WQ, q_b, nullptr, H1, TOKENS, DMODEL, DMODEL, qscale, 0);
  gemm(H0, WK, nullptr, nullptr, H2, TOKENS, DMODEL, DMODEL, 1.f, 0);
  gemm(H0, WV, v_b, nullptr, H3, TOKENS, DMODEL, DMODEL, 1.f, 0);
  attn_spatial<<<dim3(NTOK, NHEAD, BT_), dim3(256), 0, stream>>>(H1, H2, H3, BIAS, F0);
  ln(F0, attn_ln_w, attn_ln_b, nullptr, H1, TOKENS, DMODEL);
  gemm(H1, WO, out_b, F1, H2, TOKENS, DMODEL, DMODEL, 1.f, 0);              // xs -> F1 (f32) + H2 (bf16)
  gemm(H2, WS1, s_fc1_b, nullptr, H3, TOKENS, ADH, DMODEL, 1.f, 1);         // gelu fc1 -> H3
  gemm(H3, WS2, s_fc2_b, F1, nullptr, TOKENS, DMODEL, ADH, 1.f, 2);         // F1 += fc2 (skip-adapter)
  gamma_add4_kernel<<<dim3(NE4B), dim3(256), 0, stream>>>(
      (const float4*)x, gamma1, (const float4*)F1, (float4*)F2, NE4, D4);   // x2 -> F2

  // ---- final LN, MLP adapter, FFN ----
  ln(F2, fin_ln_w, fin_ln_b, nullptr, H0, TOKENS, DMODEL);                  // xn -> H0
  gemm(H0, WM1, m_fc1_b, nullptr, H3, TOKENS, ADH, DMODEL, 1.f, 1);         // gelu fc1 -> H3
  gemm(H3, WM2, m_fc2_b, F1, nullptr, TOKENS, DMODEL, ADH, 1.f, 0);         // m branch -> F1

  for (int c = 0; c < 8; ++c) {
    const unsigned short* Ac = H0 + (size_t)c * FFN_CHUNK * DMODEL;
    float* G0 = (float*)H2;     // [2056][6144] f32
    float* G1 = (float*)H3;
    unsigned short* G2 = H1;    // [2056][6144] bf16
    gemm(Ac, WI0, nullptr, G0, nullptr, FFN_CHUNK, FFN_D, DMODEL, 1.f, 0);
    gemm(Ac, WI1, nullptr, G1, nullptr, FFN_CHUNK, FFN_D, DMODEL, 1.f, 0);
    {
      const int n4 = FFN_CHUNK * FFN_D / 4;
      gelu_mul4_kernel<<<dim3((unsigned)((n4 + 255) / 256)), dim3(256), 0, stream>>>(
          (float4*)G0, (const float4*)G1, n4);
    }
    ln(G0, ffn_ln_w, ffn_ln_b, nullptr, G2, FFN_CHUNK, FFN_D);
    gemm(G2, WFO, ffn_out_b, F0 + (size_t)c * FFN_CHUNK * DMODEL, nullptr,
         FFN_CHUNK, DMODEL, FFN_D, 1.f, 0);
  }

  // out = x2 + gamma2 * ffn + m_branch
  final4_kernel<<<dim3(NE4B), dim3(256), 0, stream>>>(
      (const float4*)F2, gamma2, (const float4*)F0, (const float4*)F1, (float4*)d_out, NE4, D4);
}
